// NoSoftmaxGPT2Model_78116865179974
// MI455X (gfx1250) — matrix operand order repaired and checked by reading, not compiled, not run on hardware
//
#include <hip/hip_runtime.h>
#include <hip/hip_bf16.h>
#include <math.h>
#include <stdint.h>

// ---------------------------------------------------------------------------
// Model dims (fixed by the reference)
// ---------------------------------------------------------------------------
#define S_LEN  2048
#define EMB    768
#define NHEAD  12
#define DHEAD  64
#define FFN    3072
#define NLAYER 4
#define LN_EPS 1e-5f

// GEMM tiling: 256 threads = 8 wave32; wave grid 4x2; each wave 32x32 patch
#define BM 128
#define BN 64
#define BK 32

typedef __bf16 v16bf __attribute__((ext_vector_type(16)));
typedef float  v8f   __attribute__((ext_vector_type(8)));

union FragBF { v16bf v; int4 q[2]; };

__device__ __forceinline__ unsigned short f2bf(float f) {
    unsigned int u = __float_as_uint(f);
    u += 0x7fffu + ((u >> 16) & 1u);           // round-to-nearest-even
    return (unsigned short)(u >> 16);
}
__device__ __forceinline__ float bf2f(unsigned short h) {
    return __uint_as_float(((unsigned int)h) << 16);
}
__device__ __forceinline__ v8f zero_v8f() {
    v8f z;
#pragma unroll
    for (int i = 0; i < 8; ++i) z[i] = 0.0f;
    return z;
}

// CDNA5 async global->LDS DMA (ASYNCcnt); lds arg = LDS byte offset
// (low 32 bits of the flat address of a __shared__ object, ISA 10.2).
__device__ __forceinline__ void async_copy_b128(unsigned lds_off,
                                                unsigned long long gaddr) {
    asm volatile("global_load_async_to_lds_b128 %0, %1, off"
                 :: "v"(lds_off), "v"(gaddr) : "memory");
}
__device__ __forceinline__ void wait_async0() {
    asm volatile("s_wait_asynccnt 0x0" ::: "memory");
}

// ---------------------------------------------------------------------------
// Fused WMMA GEMM:  out = [resid +] act( A(bf16)[M,K] @ B(f32)[K,N] + bias )
// Double-buffered LDS; A staged with global_load_async_to_lds_b128; B
// converted f32->bf16 (RNE) and packed along K into b32 LDS stores.
// grid = (M/BM, N/BN, Z) with per-z element offsets for batched-head GEMMs.
// ---------------------------------------------------------------------------
template<bool GELU_ACT, bool RESID>
__global__ __launch_bounds__(256)
void gemm_bf16_kernel(const unsigned short* __restrict__ A,
                      const float*          __restrict__ B,
                      const float*          __restrict__ bias,   // may be null
                      const float*          __restrict__ resid,  // used if RESID
                      float*                __restrict__ outF,   // may be null
                      unsigned short*       __restrict__ outH,   // may be null
                      int K, int lda, int ldb, int ldc,
                      long aOffZ, long bOffZ, long cOffZ)
{
    __shared__ __align__(16) unsigned short As[2][BM * BK];  // [buf][row][k]
    __shared__ __align__(16) unsigned short Bs[2][BN * BK];  // [buf][col][k]

    const int tid  = threadIdx.x;
    const int lane = tid & 31;
    const int wave = tid >> 5;
    const int wm   = wave >> 1;          // 0..3 -> 32-row band
    const int wn   = wave & 1;           // 0..1 -> 32-col band
    const int row0 = blockIdx.x * BM;
    const int col0 = blockIdx.y * BN;

    A += (long)blockIdx.z * aOffZ;
    B += (long)blockIdx.z * bOffZ;
    const long cZ = (long)blockIdx.z * cOffZ;

    // ---- staging helpers -------------------------------------------------
    // A: 128x32 bf16 = 512 chunks of 16B; 2 async DMAs per thread.
    auto stageA_async = [&](int bsel, int k0) {
#pragma unroll
        for (int it = 0; it < 2; ++it) {
            const int c  = tid + it * 256;
            const int r  = c >> 2;
            const int kc = (c & 3) * 8;
            const unsigned lds = (unsigned)(uintptr_t)&As[bsel][r * BK + kc];
            const unsigned long long g = (unsigned long long)(uintptr_t)
                (A + (long)(row0 + r) * lda + k0 + kc);
            async_copy_b128(lds, g);
        }
    };
    // B: thread owns a 2(K) x 4(N) patch; loads now, stores later.
    const int kb = (tid >> 4) * 2;      // 0,2,..,30
    const int nb = (tid & 15) * 4;      // 0,4,..,60
    auto loadB = [&](int k0, float4& r0, float4& r1) {
        r0 = *reinterpret_cast<const float4*>(B + (long)(k0 + kb) * ldb + col0 + nb);
        r1 = *reinterpret_cast<const float4*>(B + (long)(k0 + kb + 1) * ldb + col0 + nb);
    };
    auto storeB = [&](int bsel, const float4& r0, const float4& r1) {
        const float lo[4] = {r0.x, r0.y, r0.z, r0.w};
        const float hi[4] = {r1.x, r1.y, r1.z, r1.w};
#pragma unroll
        for (int j = 0; j < 4; ++j) {   // pack (k,k+1) bf16 pair -> b32 store
            unsigned u0 = __float_as_uint(lo[j]); u0 += 0x7fffu + ((u0 >> 16) & 1u);
            unsigned u1 = __float_as_uint(hi[j]); u1 += 0x7fffu + ((u1 >> 16) & 1u);
            *reinterpret_cast<unsigned*>(&Bs[bsel][(nb + j) * BK + kb]) =
                (u0 >> 16) | (u1 & 0xffff0000u);
        }
    };

    v8f acc[2][2];
#pragma unroll
    for (int mt = 0; mt < 2; ++mt)
#pragma unroll
        for (int nt = 0; nt < 2; ++nt) acc[mt][nt] = zero_v8f();

    // ---- prologue: stage tile 0 into buffer 0 ----------------------------
    float4 rb0, rb1;
    loadB(0, rb0, rb1);
    stageA_async(0, 0);
    storeB(0, rb0, rb1);
    wait_async0();
    __syncthreads();

    // ---- main loop: 1 barrier / step, async A staging overlapped ---------
    int cur = 0;
    for (int k0 = 0; k0 < K; k0 += BK, cur ^= 1) {
        const bool has_next = (k0 + BK) < K;
        if (has_next) {
            stageA_async(cur ^ 1, k0 + BK);   // DMA into other buffer
            loadB(k0 + BK, rb0, rb1);         // B rows for next step -> regs
            if (k0 + 2 * BK < K)              // global_prefetch of step+2
                __builtin_prefetch(B + (long)(k0 + 2 * BK + kb) * ldb + col0 + nb, 0, 1);
        }

        // fragments per ISA VGPR layouts, 4 WMMAs / wave / step
        const int l15 = lane & 15;
        const int kA  = (lane < 16) ? 0 : 8;   // A: lanes0-15 K{0..7,16..23}
        const int kB  = (lane < 16) ? 0 : 8;   // B: lanes0-15 K{0..7,16..23}
        FragBF fa[2], fb[2];
#pragma unroll
        for (int mt = 0; mt < 2; ++mt) {
            const unsigned short* p = &As[cur][(wm * 32 + mt * 16 + l15) * BK + kA];
            fa[mt].q[0] = *reinterpret_cast<const int4*>(p);
            fa[mt].q[1] = *reinterpret_cast<const int4*>(p + 16);
        }
#pragma unroll
        for (int nt = 0; nt < 2; ++nt) {
            const unsigned short* p = &Bs[cur][(wn * 32 + nt * 16 + l15) * BK + kB];
            fb[nt].q[0] = *reinterpret_cast<const int4*>(p);
            fb[nt].q[1] = *reinterpret_cast<const int4*>(p + 16);
        }
#pragma unroll
        for (int mt = 0; mt < 2; ++mt)
#pragma unroll
            for (int nt = 0; nt < 2; ++nt)
                acc[mt][nt] = __builtin_amdgcn_wmma_f32_16x16x32_bf16(
                    false, fa[mt].v, false, fb[nt].v,
                    (short)0, acc[mt][nt], false, false);

        if (has_next) {
            storeB(cur ^ 1, rb0, rb1);  // finish staging next buffer
            wait_async0();              // wave's async DMAs landed
            __syncthreads();            // publish buffer cur^1
        }
    }

    // ---- epilogue: bias + gelu + residual, dual f32/bf16 store -----------
    const int l15e = lane & 15;
    const int rAdd = (lane < 16) ? 0 : 8;   // C/D: lanes16-31 hold M=8..15
#pragma unroll
    for (int mt = 0; mt < 2; ++mt) {
#pragma unroll
        for (int nt = 0; nt < 2; ++nt) {
            const int col = col0 + wn * 32 + nt * 16 + l15e;
            const float bv = bias ? bias[col] : 0.0f;
#pragma unroll
            for (int r = 0; r < 8; ++r) {
                const int row = row0 + wm * 32 + mt * 16 + rAdd + r;
                const long oidx = cZ + (long)row * ldc + col;
                float v = acc[mt][nt][r] + bv;
                if (GELU_ACT) v = 0.5f * v * (1.0f + erff(v * 0.70710678118654752f));
                if (RESID)    v += resid[oidx];
                if (outF) outF[oidx] = v;
                if (outH) outH[oidx] = f2bf(v);
            }
        }
    }
}

// ---------------------------------------------------------------------------
// LayerNorm over last dim (E=768); one block per row; bf16 and/or f32 out.
// ---------------------------------------------------------------------------
__global__ __launch_bounds__(256)
void layernorm_kernel(const float* __restrict__ x,
                      const float* __restrict__ g,
                      const float* __restrict__ b,
                      unsigned short* __restrict__ outH,
                      float* __restrict__ outF)
{
    __shared__ float red[256];
    const int row = blockIdx.x;
    const int tid = threadIdx.x;
    const float* xr = x + (long)row * EMB;

    float s = 0.0f, sq = 0.0f;
    for (int i = tid; i < EMB; i += 256) { const float v = xr[i]; s += v; sq += v * v; }
    red[tid] = s; __syncthreads();
    for (int o = 128; o > 0; o >>= 1) { if (tid < o) red[tid] += red[tid + o]; __syncthreads(); }
    const float mean = red[0] * (1.0f / EMB);
    __syncthreads();
    red[tid] = sq; __syncthreads();
    for (int o = 128; o > 0; o >>= 1) { if (tid < o) red[tid] += red[tid + o]; __syncthreads(); }
    const float var  = red[0] * (1.0f / EMB) - mean * mean;
    const float rstd = rsqrtf(var + LN_EPS);

    for (int i = tid; i < EMB; i += 256) {
        const float v = (xr[i] - mean) * rstd * g[i] + b[i];
        if (outH) outH[(long)row * EMB + i] = f2bf(v);
        if (outF) outF[(long)row * EMB + i] = v;
    }
}

// x = inputs_embeds + wpe
__global__ void add_embed_kernel(const float* __restrict__ a,
                                 const float* __restrict__ w,
                                 float* __restrict__ x, int n)
{
    const int i = blockIdx.x * blockDim.x + threadIdx.x;
    if (i < n) x[i] = a[i] + w[i];
}

// KtV[h] = scale * K_hᵀ @ V_h  (64x64 per head; no softmax => attention is
// Q @ (scale * KᵀV), associativity of the purely-linear reference).
__global__ __launch_bounds__(256)
void ktv_kernel(const unsigned short* __restrict__ kq,
                const unsigned short* __restrict__ vq,
                float* __restrict__ ktv)
{
    const int h = blockIdx.x;
    const int n = threadIdx.x & 63;
    const int m = blockIdx.y * 4 + (threadIdx.x >> 6);
    const unsigned short* kp = kq + h * DHEAD + m;
    const unsigned short* vp = vq + h * DHEAD + n;
    float acc = 0.0f;
    for (int s = 0; s < S_LEN; ++s)
        acc += bf2f(kp[(long)s * EMB]) * bf2f(vp[(long)s * EMB]);
    ktv[((long)h * DHEAD + m) * DHEAD + n] = acc * 0.125f;  // DH^-0.5 = 1/8
}

// ---------------------------------------------------------------------------
// Host launcher
// ---------------------------------------------------------------------------
extern "C" void kernel_launch(void* const* d_in, const int* in_sizes, int n_in,
                              void* d_out, int out_size, void* d_ws, size_t ws_size,
                              hipStream_t stream)
{
    const float* inputs_embeds = (const float*)d_in[0];
    const float* wpe   = (const float*)d_in[1];
    const float* ln1_g = (const float*)d_in[2];
    const float* ln1_b = (const float*)d_in[3];
    const float* Wq = (const float*)d_in[4];
    const float* bq = (const float*)d_in[5];
    const float* Wk = (const float*)d_in[6];
    const float* bk = (const float*)d_in[7];
    const float* Wv = (const float*)d_in[8];
    const float* bv = (const float*)d_in[9];
    const float* Wo = (const float*)d_in[10];
    const float* bo = (const float*)d_in[11];
    const float* ln2_g = (const float*)d_in[12];
    const float* ln2_b = (const float*)d_in[13];
    const float* W1 = (const float*)d_in[14];
    const float* b1 = (const float*)d_in[15];
    const float* W2 = (const float*)d_in[16];
    const float* b2 = (const float*)d_in[17];
    const float* lnf_g = (const float*)d_in[18];
    const float* lnf_b = (const float*)d_in[19];

    // ---- workspace carve-out (256B aligned) ----
    char* ws = (char*)d_ws;
    size_t off = 0;
    auto carve = [&](size_t bytes) -> void* {
        void* p = ws + off;
        off = (off + bytes + 255) & ~(size_t)255;
        return p;
    };
    float*          x    = (float*)carve((size_t)S_LEN * EMB * 4);
    unsigned short* hbuf = (unsigned short*)carve((size_t)S_LEN * EMB * 2);
    unsigned short* qbuf = (unsigned short*)carve((size_t)S_LEN * EMB * 2);
    unsigned short* kbuf = (unsigned short*)carve((size_t)S_LEN * EMB * 2);
    unsigned short* vbuf = (unsigned short*)carve((size_t)S_LEN * EMB * 2);
    unsigned short* abuf = (unsigned short*)carve((size_t)S_LEN * EMB * 2);
    unsigned short* m1   = (unsigned short*)carve((size_t)S_LEN * FFN * 2);
    float*          ktv  = (float*)carve((size_t)NHEAD * DHEAD * DHEAD * 4);
    (void)ws_size; (void)in_sizes; (void)n_in; (void)out_size;

    const size_t EE = (size_t)EMB * EMB;
    const size_t EF = (size_t)EMB * FFN;

    const dim3 blk(256);
    const dim3 gE (S_LEN / BM, EMB / BN, 1);      // M=2048, N=768
    const dim3 gF1(S_LEN / BM, FFN / BN, 1);      // M=2048, N=3072
    const dim3 gAp(S_LEN / BM, 1, NHEAD);         // per-head 2048x64x64

    { const int n = S_LEN * EMB;
      add_embed_kernel<<<(n + 255) / 256, 256, 0, stream>>>(inputs_embeds, wpe, x, n); }

    for (int l = 0; l < NLAYER; ++l) {
        const float* wq = Wq + l * EE; const float* bql = bq + l * EMB;
        const float* wk = Wk + l * EE; const float* bkl = bk + l * EMB;
        const float* wv = Wv + l * EE; const float* bvl = bv + l * EMB;
        const float* wo = Wo + l * EE; const float* bol = bo + l * EMB;
        const float* w1 = W1 + l * EF; const float* b1l = b1 + l * FFN;
        const float* w2 = W2 + l * EF; const float* b2l = b2 + l * EMB;

        // h = LN1(x)  (bf16 for the matrix pipes)
        layernorm_kernel<<<S_LEN, blk, 0, stream>>>(x, ln1_g + l * EMB, ln1_b + l * EMB,
                                                    hbuf, nullptr);
        // q/k/v = h @ W* + b*
        gemm_bf16_kernel<false,false><<<gE, blk, 0, stream>>>(hbuf, wq, bql, nullptr,
            nullptr, qbuf, EMB, EMB, EMB, EMB, 0, 0, 0);
        gemm_bf16_kernel<false,false><<<gE, blk, 0, stream>>>(hbuf, wk, bkl, nullptr,
            nullptr, kbuf, EMB, EMB, EMB, EMB, 0, 0, 0);
        gemm_bf16_kernel<false,false><<<gE, blk, 0, stream>>>(hbuf, wv, bvl, nullptr,
            nullptr, vbuf, EMB, EMB, EMB, EMB, 0, 0, 0);

        // KtV = scale * KᵀV per head (64x64)
        ktv_kernel<<<dim3(NHEAD, DHEAD / 4), blk, 0, stream>>>(kbuf, vbuf, ktv);

        // a[:, h] = Q_h @ KtV_h  (batched over grid.z = heads)
        gemm_bf16_kernel<false,false><<<gAp, blk, 0, stream>>>(qbuf, ktv, nullptr, nullptr,
            nullptr, abuf, DHEAD, EMB, DHEAD, EMB,
            (long)DHEAD, (long)DHEAD * DHEAD, (long)DHEAD);

        // x = x + a @ Wo + bo
        gemm_bf16_kernel<false,true><<<gE, blk, 0, stream>>>(abuf, wo, bol, x,
            x, nullptr, EMB, EMB, EMB, EMB, 0, 0, 0);

        // h = LN2(x)
        layernorm_kernel<<<S_LEN, blk, 0, stream>>>(x, ln2_g + l * EMB, ln2_b + l * EMB,
                                                    hbuf, nullptr);
        // m1 = gelu(h @ W1 + b1)
        gemm_bf16_kernel<true,false><<<gF1, blk, 0, stream>>>(hbuf, w1, b1l, nullptr,
            nullptr, m1, EMB, EMB, FFN, FFN, 0, 0, 0);
        // x = x + m1 @ W2 + b2
        gemm_bf16_kernel<false,true><<<gE, blk, 0, stream>>>(m1, w2, b2l, x,
            x, nullptr, FFN, FFN, EMB, EMB, 0, 0, 0);
    }

    // final LN -> fp32 output
    layernorm_kernel<<<S_LEN, blk, 0, stream>>>(x, lnf_g, lnf_b, nullptr, (float*)d_out);
}
